// SelectiveSSM_82609400971396
// MI455X (gfx1250) — matrix operand order repaired and checked by reading, not compiled, not run on hardware
//
#include <hip/hip_runtime.h>
#include <hip/hip_bf16.h>
#include <math.h>

// ---------------------------------------------------------------------------
// Problem constants
// ---------------------------------------------------------------------------
#define BATCH   2
#define SEQ     1024
#define DM      1024          // D_MODEL
#define DI      2048          // D_INNER
#define DXZ     4096          // 2*D_INNER
#define DS      16            // D_STATE
#define NTOK    (BATCH*SEQ)   // 2048 tokens
#define NCH     (BATCH*DI)    // 4096 scan channels
#define CHUNK   32            // scan chunk length
#define NCHUNK  (SEQ/CHUNK)   // 32 chunks

typedef __bf16  v8bf  __attribute__((ext_vector_type(8)));
typedef __bf16  v16bf __attribute__((ext_vector_type(16)));
typedef float   v8f   __attribute__((ext_vector_type(8)));

// ---------------------------------------------------------------------------
// Workspace layout (bytes).  Total ~124 MiB.
// ---------------------------------------------------------------------------
#define MB (1024ull*1024ull)
static const size_t OFF_XN_HI  = 0;              // 2048*1024 bf16  = 4 MiB
static const size_t OFF_XN_LO  = 4*MB;           // 4 MiB
static const size_t OFF_WIN_HI = 8*MB;           // Wt_in [4096][1024] bf16 = 8 MiB
static const size_t OFF_WIN_LO = 16*MB;          // 8 MiB
static const size_t OFF_WOU_HI = 24*MB;          // Wt_out [1024][2048] bf16 = 4 MiB
static const size_t OFF_WOU_LO = 28*MB;          // 4 MiB
static const size_t OFF_XZ     = 32*MB;          // 2048*4096 f32 = 32 MiB
static const size_t OFF_XS     = 64*MB;          // 2048*2048 f32 = 16 MiB
static const size_t OFF_DT     = 80*MB;          // 2048 f32
static const size_t OFF_DTB    = 81*MB;          // 2048*16 f32
static const size_t OFF_CP     = 82*MB;          // 2048*16 f32
static const size_t OFF_WXT    = 83*MB;          // Wxt [33][2048] f32 = 264 KiB
static const size_t OFF_P      = 84*MB;          // 32*4096*16 f32 = 8 MiB
static const size_t OFF_Q      = 92*MB;          // 8 MiB
static const size_t OFF_H      = 100*MB;         // 8 MiB
static const size_t OFF_G_HI   = 108*MB;         // 2048*2048 bf16 = 8 MiB
static const size_t OFF_G_LO   = 116*MB;         // 8 MiB  -> end 124 MiB

__device__ __forceinline__ float silu_f(float v) {
    return v / (1.f + __expf(-v));
}

// ---------------------------------------------------------------------------
// K1: LayerNorm -> bf16 hi/lo split (GEMM1 A-matrix)
// one block (256 thr) per token; each thread owns 4 contiguous columns
// ---------------------------------------------------------------------------
__global__ void ln_kernel(const float* __restrict__ x,
                          const float* __restrict__ lnw,
                          const float* __restrict__ lnb,
                          __bf16* __restrict__ xn_hi,
                          __bf16* __restrict__ xn_lo) {
    const int token = blockIdx.x;
    const int t = threadIdx.x;
    const float4 v  = ((const float4*)(x + (size_t)token * DM))[t];
    float s  = v.x + v.y + v.z + v.w;
    float ss = v.x*v.x + v.y*v.y + v.z*v.z + v.w*v.w;
    __shared__ float rs[256], rq[256];
    rs[t] = s; rq[t] = ss;
    __syncthreads();
    #pragma unroll
    for (int o = 128; o > 0; o >>= 1) {
        if (t < o) { rs[t] += rs[t+o]; rq[t] += rq[t+o]; }
        __syncthreads();
    }
    const float mu   = rs[0] * (1.f/DM);
    const float var  = rq[0] * (1.f/DM) - mu*mu;
    const float rstd = rsqrtf(var + 1e-5f);
    const float4 wv = ((const float4*)lnw)[t];
    const float4 bv = ((const float4*)lnb)[t];
    float o4[4] = { (v.x-mu)*rstd*wv.x + bv.x, (v.y-mu)*rstd*wv.y + bv.y,
                    (v.z-mu)*rstd*wv.z + bv.z, (v.w-mu)*rstd*wv.w + bv.w };
    const size_t base = (size_t)token * DM + 4*t;
    #pragma unroll
    for (int i = 0; i < 4; ++i) {
        __bf16 h = (__bf16)o4[i];
        xn_hi[base+i] = h;
        xn_lo[base+i] = (__bf16)(o4[i] - (float)h);
    }
}

// ---------------------------------------------------------------------------
// K2: weight convert: f32 [K][N] row-major -> bf16 hi/lo transposed [N][K]
// ---------------------------------------------------------------------------
__global__ void conv_weights_kernel(const float* __restrict__ W, int K, int N,
                                    __bf16* __restrict__ hi, __bf16* __restrict__ lo) {
    size_t i = (size_t)blockIdx.x * 256 + threadIdx.x;
    if (i >= (size_t)K * N) return;
    int k = (int)(i / N), n = (int)(i % N);
    float v = W[i];
    __bf16 h = (__bf16)v;
    hi[(size_t)n * K + k] = h;
    lo[(size_t)n * K + k] = (__bf16)(v - (float)h);
}

// K2b: transpose W_x [2048][33] -> Wxt [33][2048] (f32) for coalesced reads
__global__ void transpose_wx_kernel(const float* __restrict__ Wx,
                                    float* __restrict__ Wxt) {
    const int i = blockIdx.x * 256 + threadIdx.x;   // 33*2048 = 67584 exact
    const int k = i / 33, n = i % 33;
    Wxt[(size_t)n * DI + k] = Wx[i];
}

// ---------------------------------------------------------------------------
// WMMA fragment loaders (CDNA5 ISA §7.12.2, wave32)
// A 16x32 bf16 (row-major, leading dim ldk):
//   lanes 0-15 : row=lane,  halves[0..7]=K 0..7,  halves[8..15]=K 16..23
//   lanes 16-31: row=lane-16, halves   =K 8..15,             =K 24..31
// ---------------------------------------------------------------------------
__device__ __forceinline__ v16bf load_a_frag(const __bf16* base, int ldk,
                                             int lane, int kb) {
    const int r = lane & 15, half = lane >> 4;
    const __bf16* p = base + (size_t)r * ldk + kb + half * 8;
    v8bf a = *(const v8bf*)(p);
    v8bf b = *(const v8bf*)(p + 16);
    v16bf f;
    #pragma unroll
    for (int i = 0; i < 8; ++i) { f[i] = a[i]; f[i+8] = b[i]; }
    return f;
}
// B 32x16 bf16 from K-major weights Wt[n][k]:
//   lanes 0-15 : col n=lane,   K kb+0..7,16..23 ; lanes 16-31: col n=lane-16, K kb+8..15,24..31
__device__ __forceinline__ v16bf load_b_frag(const __bf16* base, int ldk,
                                             int lane, int kb) {
    const int n = lane & 15, half = lane >> 4;
    const __bf16* p = base + (size_t)n * ldk + kb + half * 8;
    v8bf a = *(const v8bf*)(p);
    v8bf b = *(const v8bf*)(p + 16);
    v16bf f;
    #pragma unroll
    for (int i = 0; i < 8; ++i) { f[i] = a[i]; f[i+8] = b[i]; }
    return f;
}

// ---------------------------------------------------------------------------
// K3/K9: split-precision bf16 GEMM  C[M][N] = Ahi/lo[M][K] * (Bhi/lo[N][K])^T
// one 16x16 output tile per wave, 8 waves / block, grid = (M/16*N/16)/8 exact
// 3 x v_wmma_f32_16x16x32_bf16 per K=32 step (hi*hi + hi*lo + lo*hi)
// ---------------------------------------------------------------------------
__global__ void wmma_gemm_kernel(const __bf16* __restrict__ Ahi,
                                 const __bf16* __restrict__ Alo,
                                 const __bf16* __restrict__ Bhi,
                                 const __bf16* __restrict__ Blo,
                                 float* __restrict__ C,
                                 const float* __restrict__ resid,
                                 int M, int N, int K) {
    const int wave = threadIdx.x >> 5;
    const int lane = threadIdx.x & 31;
    const int tile = blockIdx.x * 8 + wave;
    const int ntile = N >> 4;
    const int tm = tile / ntile;
    const int tn = tile % ntile;

    const __bf16* Ah = Ahi + (size_t)(tm * 16) * K;
    const __bf16* Al = Alo + (size_t)(tm * 16) * K;
    const __bf16* Bh = Bhi + (size_t)(tn * 16) * K;
    const __bf16* Bl = Blo + (size_t)(tn * 16) * K;

    v8f acc = {};
    for (int kb = 0; kb < K; kb += 32) {
        // WGP-scope prefetch of the next K tile (locality 3 -> near caches)
        __builtin_prefetch(Ah + (size_t)(lane & 15) * K + kb + 32, 0, 3);
        __builtin_prefetch(Bh + (size_t)(lane & 15) * K + kb + 32, 0, 3);
        v16bf ah = load_a_frag(Ah, K, lane, kb);
        v16bf al = load_a_frag(Al, K, lane, kb);
        v16bf bh = load_b_frag(Bh, K, lane, kb);
        v16bf bl = load_b_frag(Bl, K, lane, kb);
        acc = __builtin_amdgcn_wmma_f32_16x16x32_bf16(false, ah, false, bh,
                                                      (short)0, acc, false, false);
        acc = __builtin_amdgcn_wmma_f32_16x16x32_bf16(false, ah, false, bl,
                                                      (short)0, acc, false, false);
        acc = __builtin_amdgcn_wmma_f32_16x16x32_bf16(false, al, false, bh,
                                                      (short)0, acc, false, false);
    }
    // C/D f32 layout: VGPR r -> (M = r + 8*(lane>=16), N = lane&15)
    const int col   = tn * 16 + (lane & 15);
    const int rbase = tm * 16 + (lane >> 4) * 8;
    #pragma unroll
    for (int r = 0; r < 8; ++r) {
        size_t o = (size_t)(rbase + r) * N + col;
        float v = acc[r];
        if (resid) v += resid[o];
        C[o] = v;
    }
}

// ---------------------------------------------------------------------------
// K4: depthwise causal conv (width 4) + bias + SiLU   xz[:, :2048] -> xs
// ---------------------------------------------------------------------------
__global__ void conv_silu_kernel(const float* __restrict__ xz,
                                 const float* __restrict__ cw,
                                 const float* __restrict__ cb,
                                 float* __restrict__ xs) {
    const size_t idx = (size_t)blockIdx.x * 256 + threadIdx.x;  // 2^22 total
    const int d = (int)(idx & (DI - 1));
    const int l = (int)((idx >> 11) & (SEQ - 1));
    const int b = (int)(idx >> 21);
    const size_t base = ((size_t)b * SEQ) * DXZ + d;
    float acc = cb[d];
    const float* w = cw + d * 4;
    #pragma unroll
    for (int j = 0; j < 4; ++j) {
        const int ls = l - 3 + j;
        if (ls >= 0) acc += w[j] * xz[base + (size_t)ls * DXZ];
    }
    xs[((size_t)b * SEQ + l) * DI + d] = silu_f(acc);
}

// ---------------------------------------------------------------------------
// K5: skinny GEMM  xp = xs @ W_x  (N=33) using transposed Wxt for coalescing.
// Emits dt = softplus(xp[32]); dtB[n] = dt*xp[n]; Cp[n] = xp[16+n].
// ---------------------------------------------------------------------------
__global__ void xp_kernel(const float* __restrict__ xs,
                          const float* __restrict__ Wxt,   // [33][2048]
                          float* __restrict__ dt,
                          float* __restrict__ dtB,
                          float* __restrict__ Cp) {
    const int token = blockIdx.x;
    const int t = threadIdx.x;
    const float* xr = xs + (size_t)token * DI;
    float xv[8];
    #pragma unroll
    for (int i = 0; i < 8; ++i) xv[i] = xr[t + i * 256];
    float acc[33];
    #pragma unroll
    for (int n = 0; n < 33; ++n) {
        const float* wrow = Wxt + (size_t)n * DI;
        float a = 0.f;
        #pragma unroll
        for (int i = 0; i < 8; ++i) a += xv[i] * wrow[t + i * 256];
        acc[n] = a;
    }
    __shared__ float part[33 * 8];
    __shared__ float xps[33];
    const int lane = t & 31, wv = t >> 5;
    #pragma unroll
    for (int n = 0; n < 33; ++n) {
        float v = acc[n];
        #pragma unroll
        for (int off = 16; off > 0; off >>= 1) v += __shfl_down(v, off, 32);
        if (lane == 0) part[n * 8 + wv] = v;
    }
    __syncthreads();
    if (t < 33) {
        float v = 0.f;
        #pragma unroll
        for (int i = 0; i < 8; ++i) v += part[t * 8 + i];
        xps[t] = v;
    }
    __syncthreads();
    if (t < 16) {
        const float dv  = xps[32];
        const float dlt = (dv > 20.f) ? dv : log1pf(__expf(dv));
        dtB[token * 16 + t] = dlt * xps[t];
        Cp [token * 16 + t] = xps[16 + t];
        if (t == 0) dt[token] = dlt;
    }
}

// ---------------------------------------------------------------------------
// Chunked associative scan: h_t = a_t*h + u_t,  a_t[n]=exp(dt*A[d][n]),
// u_t[n]=dtB_t[n]*x_t[d].
// Pass 1: per-(channel,chunk) transition (P = prod a, Q = run from h=0)
// ---------------------------------------------------------------------------
__global__ void scan_pass1_kernel(const float* __restrict__ xs,
                                  const float* __restrict__ dt,
                                  const float* __restrict__ dtB,
                                  const float* __restrict__ logA,
                                  float* __restrict__ P,
                                  float* __restrict__ Q) {
    const int idx = blockIdx.x * 256 + threadIdx.x;   // NCH * NCHUNK
    const int j = idx >> 12;           // chunk
    const int c = idx & (NCH - 1);     // channel = b*DI + d
    const int b = c >> 11, d = c & (DI - 1);
    float A[DS], h[DS], p[DS];
    #pragma unroll
    for (int n = 0; n < DS; ++n) {
        A[n] = -__expf(logA[d * DS + n]);
        h[n] = 0.f; p[n] = 1.f;
    }
    const int t0 = j * CHUNK;
    for (int s = 0; s < CHUNK; ++s) {
        const int token = b * SEQ + t0 + s;
        const float dlt = dt[token];
        const float xv  = xs[(size_t)token * DI + d];
        #pragma unroll
        for (int n = 0; n < DS; ++n) {
            const float e = __expf(dlt * A[n]);
            h[n] = e * h[n] + dtB[token * 16 + n] * xv;
            p[n] *= e;
        }
    }
    const size_t o = ((size_t)j * NCH + c) * DS;
    #pragma unroll
    for (int n = 0; n < DS; ++n) { P[o + n] = p[n]; Q[o + n] = h[n]; }
}

// Pass 2: 32-step scan across chunks; H = per-chunk initial states
__global__ void scan_pass2_kernel(const float* __restrict__ init,
                                  const float* __restrict__ P,
                                  const float* __restrict__ Q,
                                  float* __restrict__ H) {
    const int idx = blockIdx.x * 256 + threadIdx.x;   // NCH*DS = 65536 = c*16+n
    float h = init[idx];
    for (int j = 0; j < NCHUNK; ++j) {
        const size_t o = (size_t)j * (NCH * DS) + idx;
        H[o] = h;
        h = P[o] * h + Q[o];
    }
}

// Pass 3: replay within chunk, y = <h,C> + D*x, g = y*silu(z) -> bf16 hi/lo
__global__ void scan_pass3_kernel(const float* __restrict__ xs,
                                  const float* __restrict__ xz,
                                  const float* __restrict__ dt,
                                  const float* __restrict__ dtB,
                                  const float* __restrict__ Cp,
                                  const float* __restrict__ logA,
                                  const float* __restrict__ Dp,
                                  const float* __restrict__ H,
                                  __bf16* __restrict__ g_hi,
                                  __bf16* __restrict__ g_lo) {
    const int idx = blockIdx.x * 256 + threadIdx.x;
    const int j = idx >> 12;
    const int c = idx & (NCH - 1);
    const int b = c >> 11, d = c & (DI - 1);
    float A[DS], h[DS];
    #pragma unroll
    for (int n = 0; n < DS; ++n) {
        A[n] = -__expf(logA[d * DS + n]);
        h[n] = H[(size_t)j * (NCH * DS) + (size_t)c * DS + n];
    }
    const float Dd = Dp[d];
    const int t0 = j * CHUNK;
    for (int s = 0; s < CHUNK; ++s) {
        const int token = b * SEQ + t0 + s;
        const float dlt = dt[token];
        const float xv  = xs[(size_t)token * DI + d];
        float y = Dd * xv;
        #pragma unroll
        for (int n = 0; n < DS; ++n) {
            const float e = __expf(dlt * A[n]);
            h[n] = e * h[n] + dtB[token * 16 + n] * xv;
            y += h[n] * Cp[token * 16 + n];
        }
        const float zv = xz[(size_t)token * DXZ + DI + d];
        const float g  = y * silu_f(zv);
        const __bf16 gh = (__bf16)g;
        const size_t o = (size_t)token * DI + d;
        g_hi[o] = gh;
        g_lo[o] = (__bf16)(g - (float)gh);
    }
}

// ---------------------------------------------------------------------------
// kernel_launch
// inputs: 0:x 1:init_state 2:ln_w 3:ln_b 4:W_in 5:conv_w 6:conv_b 7:W_x
//         8:log_A 9:D_param 10:W_out
// ---------------------------------------------------------------------------
extern "C" void kernel_launch(void* const* d_in, const int* in_sizes, int n_in,
                              void* d_out, int out_size, void* d_ws, size_t ws_size,
                              hipStream_t stream) {
    (void)in_sizes; (void)n_in; (void)out_size; (void)ws_size;
    const float* x      = (const float*)d_in[0];
    const float* initst = (const float*)d_in[1];
    const float* ln_w   = (const float*)d_in[2];
    const float* ln_b   = (const float*)d_in[3];
    const float* W_in   = (const float*)d_in[4];
    const float* conv_w = (const float*)d_in[5];
    const float* conv_b = (const float*)d_in[6];
    const float* W_x    = (const float*)d_in[7];
    const float* log_A  = (const float*)d_in[8];
    const float* D_par  = (const float*)d_in[9];
    const float* W_out  = (const float*)d_in[10];
    float* out = (float*)d_out;

    char* ws = (char*)d_ws;
    __bf16* xn_hi  = (__bf16*)(ws + OFF_XN_HI);
    __bf16* xn_lo  = (__bf16*)(ws + OFF_XN_LO);
    __bf16* win_hi = (__bf16*)(ws + OFF_WIN_HI);
    __bf16* win_lo = (__bf16*)(ws + OFF_WIN_LO);
    __bf16* wou_hi = (__bf16*)(ws + OFF_WOU_HI);
    __bf16* wou_lo = (__bf16*)(ws + OFF_WOU_LO);
    float*  xz     = (float*) (ws + OFF_XZ);
    float*  xs     = (float*) (ws + OFF_XS);
    float*  dt     = (float*) (ws + OFF_DT);
    float*  dtB    = (float*) (ws + OFF_DTB);
    float*  Cp     = (float*) (ws + OFF_CP);
    float*  Wxt    = (float*) (ws + OFF_WXT);
    float*  P      = (float*) (ws + OFF_P);
    float*  Q      = (float*) (ws + OFF_Q);
    float*  H      = (float*) (ws + OFF_H);
    __bf16* g_hi   = (__bf16*)(ws + OFF_G_HI);
    __bf16* g_lo   = (__bf16*)(ws + OFF_G_LO);

    // K1: layernorm + bf16 split
    ln_kernel<<<NTOK, 256, 0, stream>>>(x, ln_w, ln_b, xn_hi, xn_lo);

    // K2: weight transpose+split (W_in: [1024][4096], W_out: [2048][1024])
    conv_weights_kernel<<<(DM * DXZ) / 256, 256, 0, stream>>>(W_in, DM, DXZ, win_hi, win_lo);
    conv_weights_kernel<<<(DI * DM) / 256, 256, 0, stream>>>(W_out, DI, DM, wou_hi, wou_lo);
    transpose_wx_kernel<<<(DI * 33) / 256, 256, 0, stream>>>(W_x, Wxt);

    // K3: GEMM1  xz[2048][4096] = xn[2048][1024] @ W_in
    wmma_gemm_kernel<<<(NTOK/16) * (DXZ/16) / 8, 256, 0, stream>>>(
        xn_hi, xn_lo, win_hi, win_lo, xz, nullptr, NTOK, DXZ, DM);

    // K4: depthwise causal conv + SiLU -> xs[2048][2048]
    conv_silu_kernel<<<(BATCH * SEQ * DI) / 256, 256, 0, stream>>>(xz, conv_w, conv_b, xs);

    // K5: xp = xs @ W_x -> dt, dtB, Cp
    xp_kernel<<<NTOK, 256, 0, stream>>>(xs, Wxt, dt, dtB, Cp);

    // K6-K8: chunked selective scan
    scan_pass1_kernel<<<(NCH * NCHUNK) / 256, 256, 0, stream>>>(xs, dt, dtB, log_A, P, Q);
    scan_pass2_kernel<<<(NCH * DS) / 256, 256, 0, stream>>>(initst, P, Q, H);
    scan_pass3_kernel<<<(NCH * NCHUNK) / 256, 256, 0, stream>>>(
        xs, xz, dt, dtB, Cp, log_A, D_par, H, g_hi, g_lo);

    // K9: GEMM3  out = residual + g[2048][2048] @ W_out
    wmma_gemm_kernel<<<(NTOK/16) * (DM/16) / 8, 256, 0, stream>>>(
        g_hi, g_lo, wou_hi, wou_lo, out, x, NTOK, DM, DI);
}
